// Pspaformer_54073638257180
// MI455X (gfx1250) — matrix operand order repaired and checked by reading, not compiled, not run on hardware
//
#include <hip/hip_runtime.h>
#include <stdint.h>

// ---------------------------------------------------------------------------
// Problem constants (from reference)
#define NN   262144      // nodes
#define CC   200         // spectral channels
#define CPAD 224         // 200 padded to multiple of 32 (bf16 WMMA K-step)
#define HID  64
#define EE   2097152     // edges
#define LL   2
#define CLS  16
#define EPSV 1e-5f

// LDS pitches chosen for 16-distinct-bank fragment loads + 16B alignment
#define BPITCH_STEM 232   // stem B tile: [64 cols][232 k]   (14848 bf16)
#define APITCH_STEM 40    // stem A tile: [128 rows][40 k]   (5120 bf16)
#define BPITCH_HID  72    // hidden B tile: [64 cols][72 k]  (4608 bf16)
#define APITCH_HID  72    // hidden A tile: [128 rows][72 k] (9216 bf16)

typedef __attribute__((ext_vector_type(16))) __bf16 v16bf;
typedef __attribute__((ext_vector_type(8)))  __bf16 bf16x8;
typedef __attribute__((ext_vector_type(4)))  __bf16 bf16x4;
typedef __attribute__((ext_vector_type(8)))  float  v8f;

// CDNA5 async copy: 16 bytes global -> LDS, tracked by ASYNCcnt (no VGPR data).
__device__ __forceinline__ void async_copy_b128(unsigned lds_off, unsigned g_off,
                                                const void* base) {
  asm volatile("global_load_async_to_lds_b128 %0, %1, %2"
               :: "v"(lds_off), "v"(g_off), "s"(base) : "memory");
}
__device__ __forceinline__ void wait_async0() {
  asm volatile("s_wait_asynccnt 0x0" ::: "memory");
}

__device__ __forceinline__ v16bf frag_cat(bf16x8 lo, bf16x8 hi) {
  return __builtin_shufflevector(lo, hi, 0, 1, 2, 3, 4, 5, 6, 7,
                                 8, 9, 10, 11, 12, 13, 14, 15);
}

// ---------------------------------------------------------------------------
// One-time weight conversion to bf16, written K-major per output column
// ("[n][k]" with padded pitch) so WMMA B fragments are contiguous 32B runs.
__global__ __launch_bounds__(256) void prep_weights(
    const float* __restrict__ preW, const float* __restrict__ wrv,
    const float* __restrict__ wcv,
    __bf16* __restrict__ bpre, __bf16* __restrict__ brw, __bf16* __restrict__ bcw) {
  int tid = blockIdx.x * blockDim.x + threadIdx.x;
  int stride = gridDim.x * blockDim.x;
  for (int i = tid; i < HID * CPAD; i += stride) {
    int n = i / CPAD, k = i % CPAD;
    bpre[n * BPITCH_STEM + k] = (__bf16)(k < CC ? preW[k * HID + n] : 0.0f);
  }
  for (int i = tid; i < LL * HID * HID; i += stride) {
    int l = i >> 12, rem = i & 4095;
    int n = rem >> 6, k = rem & 63;
    int di = l * (HID * BPITCH_HID) + n * BPITCH_HID + k;
    brw[di] = (__bf16)wrv[i];
    bcw[di] = (__bf16)wcv[i];
  }
}

// ---------------------------------------------------------------------------
// Stem GEMM: out[N,64] = x[N,200] @ W[200,64] + bias  (bf16 WMMA, f32 acc)
__global__ __launch_bounds__(256) void stem_gemm(
    const float* __restrict__ x, const __bf16* __restrict__ Wb,
    const float* __restrict__ bias, float* __restrict__ out) {
  __shared__ __align__(16) __bf16 sB[HID * BPITCH_STEM];  // ~29 KB, K-major
  __shared__ __align__(16) __bf16 sA[128 * APITCH_STEM];  // ~10 KB
  const int rowbase = blockIdx.x * 128;

  // Async-stage the whole weight tile (pure byte copy, ASYNCcnt path).
  {
    const unsigned lbase = (unsigned)(uintptr_t)&sB[0];
#pragma unroll
    for (int it = 0; it < 8; ++it) {
      int i = threadIdx.x + it * 256;                 // 16B chunk index
      if (i < (HID * BPITCH_STEM * 2) / 16)           // 1856 chunks
        async_copy_b128(lbase + i * 16, (unsigned)(i * 16), Wb);
    }
    wait_async0();
  }

  const int lane = threadIdx.x & 31;
  const int wave = threadIdx.x >> 5;
  const int half = lane >> 4;
  const int l16  = lane & 15;
  const int m0   = wave * 16;

  v8f acc[4];
#pragma unroll
  for (int ct = 0; ct < 4; ++ct) acc[ct] = (v8f){0, 0, 0, 0, 0, 0, 0, 0};

  for (int kc = 0; kc < CPAD / 32; ++kc) {
    __syncthreads();
    // stage 128x32 A chunk: float4 global loads -> packed bf16x4 LDS stores
#pragma unroll
    for (int it = 0; it < 4; ++it) {
      int g = threadIdx.x + it * 256;                 // 1024 groups of 4
      int r = g >> 3;
      int k4 = (g & 7) * 4;
      int gk = kc * 32 + k4;
      float4 v = make_float4(0.f, 0.f, 0.f, 0.f);
      if (gk + 4 <= CC)
        v = *(const float4*)(x + (size_t)(rowbase + r) * CC + gk);
      bf16x4 p;
      p[0] = (__bf16)v.x; p[1] = (__bf16)v.y; p[2] = (__bf16)v.z; p[3] = (__bf16)v.w;
      *(bf16x4*)(sA + r * APITCH_STEM + k4) = p;
    }
    __syncthreads();
    // A fragment: one contiguous 32B run, k = 16*half + s (matches B)
    const __bf16* ar = sA + (m0 + l16) * APITCH_STEM;
    v16bf a = frag_cat(*(const bf16x8*)(ar + half * 16),
                       *(const bf16x8*)(ar + half * 16 + 8));
#pragma unroll
    for (int ct = 0; ct < 4; ++ct) {
      const __bf16* br = sB + (ct * 16 + l16) * BPITCH_STEM + kc * 32 + half * 16;
      v16bf b = frag_cat(*(const bf16x8*)br, *(const bf16x8*)(br + 8));
      acc[ct] = __builtin_amdgcn_wmma_f32_16x16x32_bf16(
          false, a, false, b, (short)0, acc[ct], false, false);
    }
  }
#pragma unroll
  for (int ct = 0; ct < 4; ++ct) {
    int gc = ct * 16 + l16;
    float bv = bias[gc];
#pragma unroll
    for (int r = 0; r < 8; ++r) {
      int gr = rowbase + m0 + r + 8 * half;
      out[(size_t)gr * HID + gc] = acc[ct][r] + bv;
    }
  }
}

// ---------------------------------------------------------------------------
// Per-channel BN partial sums (sum, sumsq) -> stats[0..63], stats[64..127]
__global__ __launch_bounds__(256) void bn_stats(
    const float* __restrict__ y, float* __restrict__ stats) {
  int c = threadIdx.x & 63;
  int row = blockIdx.x * 4 + (threadIdx.x >> 6);
  int stride = gridDim.x * 4;
  float s = 0.f, sq = 0.f;
  for (int r = row; r < NN; r += stride) {
    float v = y[(size_t)r * HID + c];
    s += v; sq += v * v;
  }
  atomicAdd(&stats[c], s);
  atomicAdd(&stats[64 + c], sq);
}

// stats[128..191] = scale, stats[192..255] = shift
__global__ void bn_finalize(const float* __restrict__ g, const float* __restrict__ b,
                            float* __restrict__ stats) {
  int c = threadIdx.x;   // 64 threads
  float m = stats[c] * (1.0f / NN);
  float var = stats[64 + c] * (1.0f / NN) - m * m;   // biased var, like jnp.var
  float sc = g[c] * rsqrtf(var + EPSV);
  stats[128 + c] = sc;
  stats[192 + c] = b[c] - m * sc;
}

__global__ __launch_bounds__(256) void bn_apply(
    const float* __restrict__ y, const float* __restrict__ stats,
    float* __restrict__ out, int leaky) {
  size_t i4 = ((size_t)blockIdx.x * 256 + threadIdx.x) * 4;
  int c = (int)(i4 & 63);
  float4 v = *(const float4*)(y + i4);
  float4 r;
  r.x = v.x * stats[128 + c]     + stats[192 + c];
  r.y = v.y * stats[128 + c + 1] + stats[192 + c + 1];
  r.z = v.z * stats[128 + c + 2] + stats[192 + c + 2];
  r.w = v.w * stats[128 + c + 3] + stats[192 + c + 3];
  if (leaky) {
    r.x = (r.x > 0.f) ? r.x : 0.01f * r.x;
    r.y = (r.y > 0.f) ? r.y : 0.01f * r.y;
    r.z = (r.z > 0.f) ? r.z : 0.01f * r.z;
    r.w = (r.w > 0.f) ? r.w : 0.01f * r.w;
  }
  *(float4*)(out + i4) = r;
}

// ---------------------------------------------------------------------------
// LayerNorm + dual GEMM: xn = LN(h); rowv = xn@Wr + br; colv = xn@Wc + bc.
// Also writes xn (f32) into the residual accumulator (initializes it).
__global__ __launch_bounds__(256) void ln_gemm(
    const float* __restrict__ h,
    const float* __restrict__ lng, const float* __restrict__ lnb,
    const __bf16* __restrict__ Wr, const float* __restrict__ br,
    const __bf16* __restrict__ Wc, const float* __restrict__ bc,
    float* __restrict__ xn_out, float* __restrict__ rowv, float* __restrict__ colv) {
  __shared__ __align__(16) __bf16 sA[128 * APITCH_HID];   // 18 KB
  __shared__ __align__(16) __bf16 sBr[HID * BPITCH_HID];  // 9 KB, K-major
  __shared__ __align__(16) __bf16 sBc[HID * BPITCH_HID];  // 9 KB, K-major
  const int rowbase = blockIdx.x * 128;

  // Async-stage both weight tiles (pure byte copies).
  {
    const unsigned lr = (unsigned)(uintptr_t)&sBr[0];
    const unsigned lc = (unsigned)(uintptr_t)&sBc[0];
    const int nchunk = (HID * BPITCH_HID * 2) / 16;     // 576
#pragma unroll
    for (int it = 0; it < 3; ++it) {
      int i = threadIdx.x + it * 256;
      if (i < nchunk) {
        async_copy_b128(lr + i * 16, (unsigned)(i * 16), Wr);
        async_copy_b128(lc + i * 16, (unsigned)(i * 16), Wc);
      }
    }
    wait_async0();
  }

  if (threadIdx.x < 128) {
    const int r = rowbase + threadIdx.x;
    const float4* hr4 = (const float4*)(h + (size_t)r * HID);
    float s = 0.f, sq = 0.f;
#pragma unroll
    for (int kk = 0; kk < 16; ++kk) {
      float4 v = hr4[kk];
      s += v.x + v.y + v.z + v.w;
      sq += v.x * v.x + v.y * v.y + v.z * v.z + v.w * v.w;
    }
    float m = s * (1.0f / HID);
    float var = sq * (1.0f / HID) - m * m;
    float inv = rsqrtf(var + EPSV);
    float4* xr4 = (float4*)(xn_out + (size_t)r * HID);
    __bf16* sa = sA + threadIdx.x * APITCH_HID;
#pragma unroll
    for (int kk = 0; kk < 16; ++kk) {
      float4 v = hr4[kk];
      float4 g = *(const float4*)(lng + kk * 4);
      float4 bb = *(const float4*)(lnb + kk * 4);
      float4 o;
      o.x = (v.x - m) * inv * g.x + bb.x;
      o.y = (v.y - m) * inv * g.y + bb.y;
      o.z = (v.z - m) * inv * g.z + bb.z;
      o.w = (v.w - m) * inv * g.w + bb.w;
      xr4[kk] = o;                          // residual / accumulator init
      bf16x4 p;
      p[0] = (__bf16)o.x; p[1] = (__bf16)o.y; p[2] = (__bf16)o.z; p[3] = (__bf16)o.w;
      *(bf16x4*)(sa + kk * 4) = p;
    }
  }
  __syncthreads();

  const int lane = threadIdx.x & 31;
  const int wave = threadIdx.x >> 5;
  const int half = lane >> 4;
  const int l16  = lane & 15;
  const int m0   = wave * 16;

  v8f ar[4], ac[4];
#pragma unroll
  for (int ct = 0; ct < 4; ++ct) { ar[ct] = (v8f){0,0,0,0,0,0,0,0}; ac[ct] = ar[ct]; }

#pragma unroll
  for (int kc = 0; kc < 2; ++kc) {
    const __bf16* arow = sA + (m0 + l16) * APITCH_HID + kc * 32;
    v16bf a = frag_cat(*(const bf16x8*)(arow + half * 16),
                       *(const bf16x8*)(arow + half * 16 + 8));
#pragma unroll
    for (int ct = 0; ct < 4; ++ct) {
      const __bf16* b1p = sBr + (ct * 16 + l16) * BPITCH_HID + kc * 32 + half * 16;
      const __bf16* b2p = sBc + (ct * 16 + l16) * BPITCH_HID + kc * 32 + half * 16;
      v16bf b1 = frag_cat(*(const bf16x8*)b1p, *(const bf16x8*)(b1p + 8));
      v16bf b2 = frag_cat(*(const bf16x8*)b2p, *(const bf16x8*)(b2p + 8));
      ar[ct] = __builtin_amdgcn_wmma_f32_16x16x32_bf16(
          false, a, false, b1, (short)0, ar[ct], false, false);
      ac[ct] = __builtin_amdgcn_wmma_f32_16x16x32_bf16(
          false, a, false, b2, (short)0, ac[ct], false, false);
    }
  }
#pragma unroll
  for (int ct = 0; ct < 4; ++ct) {
    int gc = ct * 16 + l16;
    float bvr = br[gc], bvc = bc[gc];
#pragma unroll
    for (int r = 0; r < 8; ++r) {
      int gr = rowbase + m0 + r + 8 * half;
      rowv[(size_t)gr * HID + gc] = ar[ct][r] + bvr;
      colv[(size_t)gr * HID + gc] = ac[ct][r] + bvc;
    }
  }
}

// ---------------------------------------------------------------------------
// Edge aggregation: accum[src[e]] += v[dst[e]] * att[e].
// One edge per wave (uniform e -> scalar loads; coalesced gather + atomics).
// Value tables (67 MB) are L2-resident on MI455X (192 MB L2).
__global__ __launch_bounds__(256) void agg_kernel(
    const float* __restrict__ v, const float* __restrict__ att,
    const int* __restrict__ src, const int* __restrict__ dst,
    float* __restrict__ accum) {
  long gid = (long)blockIdx.x * 256 + threadIdx.x;   // EE*32 threads
  int e = (int)(gid >> 5);
  int c = (int)(gid & 31) * 2;
  float a = att[e];
  int d = dst[e], s = src[e];
  const float* vp = v + (size_t)d * HID + c;
  float v0 = vp[0], v1 = vp[1];
  float* ap = accum + (size_t)s * HID + c;
  atomicAdd(ap,     v0 * a);
  atomicAdd(ap + 1, v1 * a);
}

// ---------------------------------------------------------------------------
// Head: logits = h @ smW + smb; softmax over 16 classes. One row per thread.
__global__ __launch_bounds__(256) void head_softmax(
    const float* __restrict__ h, const float* __restrict__ W,
    const float* __restrict__ b, float* __restrict__ out) {
  __shared__ float sW[HID * CLS];
  __shared__ float sb[CLS];
  for (int i = threadIdx.x; i < HID * CLS; i += 256) sW[i] = W[i];
  if (threadIdx.x < CLS) sb[threadIdx.x] = b[threadIdx.x];
  __syncthreads();
  int row = blockIdx.x * 256 + threadIdx.x;
  float lg[CLS];
#pragma unroll
  for (int c = 0; c < CLS; ++c) lg[c] = sb[c];
  const float4* hr4 = (const float4*)(h + (size_t)row * HID);
#pragma unroll
  for (int kk = 0; kk < 16; ++kk) {
    float4 v = hr4[kk];
    const float* w0 = sW + kk * 4 * CLS;
#pragma unroll
    for (int c = 0; c < CLS; ++c)
      lg[c] += v.x * w0[c] + v.y * w0[CLS + c] + v.z * w0[2 * CLS + c] +
               v.w * w0[3 * CLS + c];
  }
  float mx = lg[0];
#pragma unroll
  for (int c = 1; c < CLS; ++c) mx = fmaxf(mx, lg[c]);
  float s = 0.f;
#pragma unroll
  for (int c = 0; c < CLS; ++c) { lg[c] = __expf(lg[c] - mx); s += lg[c]; }
  float inv = 1.0f / s;
  float* o = out + (size_t)row * CLS;
#pragma unroll
  for (int c = 0; c < CLS; ++c) o[c] = lg[c] * inv;
}

// ---------------------------------------------------------------------------
extern "C" void kernel_launch(void* const* d_in, const int* in_sizes, int n_in,
                              void* d_out, int out_size, void* d_ws, size_t ws_size,
                              hipStream_t stream) {
  const float* x       = (const float*)d_in[0];
  const float* row_att = (const float*)d_in[1];
  const float* col_att = (const float*)d_in[2];
  const float* preW    = (const float*)d_in[3];
  const float* preB    = (const float*)d_in[4];
  const float* bn0g    = (const float*)d_in[5];
  const float* bn0b    = (const float*)d_in[6];
  const float* lng     = (const float*)d_in[7];
  const float* lnb     = (const float*)d_in[8];
  const float* Wrv     = (const float*)d_in[9];
  const float* brv     = (const float*)d_in[10];
  const float* Wcv     = (const float*)d_in[11];
  const float* bcv     = (const float*)d_in[12];
  const float* bng     = (const float*)d_in[13];
  const float* bnb     = (const float*)d_in[14];
  const float* smW     = (const float*)d_in[15];
  const float* smb     = (const float*)d_in[16];
  const int* rowsrc    = (const int*)d_in[17];
  const int* rowdst    = (const int*)d_in[18];
  const int* colsrc    = (const int*)d_in[19];
  const int* coldst    = (const int*)d_in[20];
  float* out = (float*)d_out;

  // Workspace carve-out (~268 MB + small)
  float* h     = (float*)d_ws;
  float* accum = h     + (size_t)NN * HID;
  float* rowv  = accum + (size_t)NN * HID;
  float* colv  = rowv  + (size_t)NN * HID;
  float* stats = colv  + (size_t)NN * HID;        // 256 floats
  __bf16* bpre = (__bf16*)(stats + 256);          // [64][232]
  __bf16* brw  = bpre + HID * BPITCH_STEM;        // [L][64][72]
  __bf16* bcw  = brw  + LL * HID * BPITCH_HID;    // [L][64][72]

  prep_weights<<<64, 256, 0, stream>>>(preW, Wrv, Wcv, bpre, brw, bcw);

  // stem + BN
  stem_gemm<<<NN / 128, 256, 0, stream>>>(x, bpre, preB, h);
  hipMemsetAsync(stats, 0, 128 * sizeof(float), stream);
  bn_stats<<<1024, 256, 0, stream>>>(h, stats);
  bn_finalize<<<1, 64, 0, stream>>>(bn0g, bn0b, stats);
  bn_apply<<<(NN * HID) / 1024, 256, 0, stream>>>(h, stats, h, 0);

  for (int l = 0; l < LL; ++l) {
    ln_gemm<<<NN / 128, 256, 0, stream>>>(
        h, lng + l * HID, lnb + l * HID,
        brw + (size_t)l * HID * BPITCH_HID, brv + l * HID,
        bcw + (size_t)l * HID * BPITCH_HID, bcv + l * HID,
        accum, rowv, colv);
    agg_kernel<<<(EE * 32) / 256, 256, 0, stream>>>(rowv, row_att, rowsrc, rowdst, accum);
    agg_kernel<<<(EE * 32) / 256, 256, 0, stream>>>(colv, col_att, colsrc, coldst, accum);
    hipMemsetAsync(stats, 0, 128 * sizeof(float), stream);
    bn_stats<<<1024, 256, 0, stream>>>(accum, stats);
    bn_finalize<<<1, 64, 0, stream>>>(bng + l * HID, bnb + l * HID, stats);
    bn_apply<<<(NN * HID) / 1024, 256, 0, stream>>>(accum, stats, h, 1);
  }

  head_softmax<<<NN / 256, 256, 0, stream>>>(h, smW, smb, out);
}
